// MultiHeadAttention_56178172232240
// MI455X (gfx1250) — compile-verified, matrix operand order checked by reading, not run on hardware
//
#include <hip/hip_runtime.h>
#include <hip/hip_bf16.h>

// ---------------------------------------------------------------------------
// MHA forward for B=2, S=2048, D=1024, H=16, HD=64 on gfx1250 (MI455X).
// Compute-bound (~103 GFLOP vs ~50MB traffic) -> all GEMMs + attention via
// v_wmma_f32_16x16x32_bf16 with fp32 accumulation. Needs >= 48 MiB of d_ws.
// ---------------------------------------------------------------------------

typedef __attribute__((ext_vector_type(16))) __bf16 v16bf;
typedef __attribute__((ext_vector_type(8)))  float  v8f;
typedef __attribute__((ext_vector_type(4)))  int    v4i;

#define BB   2
#define SS   2048
#define DD   1024
#define HH   16
#define HD   64
#define SM_SCALE 0.125f   // 1/sqrt(64)

// CDNA5 async global->LDS path (guarded; falls back to sync copies).
#if defined(__has_builtin)
#if __has_builtin(__builtin_amdgcn_global_load_async_to_lds_b128) && \
    __has_builtin(__builtin_amdgcn_s_wait_asynccnt)
#define HAVE_ASYNC_LDS 1
#endif
#endif

#ifdef HAVE_ASYNC_LDS
typedef __attribute__((address_space(1))) v4i* as1_v4i_p;
typedef __attribute__((address_space(3))) v4i* as3_v4i_p;
#endif

__device__ __forceinline__ unsigned short f2bf(float f) {
  union { float f; unsigned u; } v; v.f = f;
  unsigned r = (v.u + 0x7FFFu + ((v.u >> 16) & 1u)) >> 16;
  return (unsigned short)r;
}

union FragU { uint4 u[2]; v16bf v; };

// Load a 16-bit WMMA A/B fragment for one lane from a row-major source.
// p points at (row base + k0 + base) where base = 8*(lane>=16).
// halves[0..7] = k base..base+7, halves[8..15] = k base+16..base+23.
__device__ __forceinline__ v16bf load_frag(const unsigned short* p) {
  FragU f;
  f.u[0] = *(const uint4*)(p);
  f.u[1] = *(const uint4*)(p + 16);
  return f.v;
}

#define WMMA_BF16(a, b, c) \
  __builtin_amdgcn_wmma_f32_16x16x32_bf16(false, (a), false, (b), (short)0, (c), false, false)

// 16-byte global -> LDS copy; async (ASYNCcnt-tracked) path when available.
__device__ __forceinline__ void copy16_g2l(const unsigned short* g,
                                           unsigned short* l) {
#ifdef HAVE_ASYNC_LDS
  // Cast through integers to sidestep const/addrspace conversion rules.
  // addrspace(3) pointers are 32-bit LDS offsets (= low 32 bits of the
  // generic shared-aperture address).
  __builtin_amdgcn_global_load_async_to_lds_b128(
      (as1_v4i_p)(size_t)g, (as3_v4i_p)(unsigned)(size_t)l, 0, 0);
#else
  *(uint4*)l = *(const uint4*)g;
#endif
}

__device__ __forceinline__ void wait_g2l() {
#ifdef HAVE_ASYNC_LDS
  __builtin_amdgcn_s_wait_asynccnt(0);
#endif
}

// ---------------------------------------------------------------------------
// fp32 -> bf16 elementwise convert
// ---------------------------------------------------------------------------
__global__ __launch_bounds__(256) void cvt_f32_bf16(
    const float* __restrict__ in, unsigned short* __restrict__ out, int n) {
  for (int i = blockIdx.x * blockDim.x + threadIdx.x; i < n;
       i += gridDim.x * blockDim.x)
    out[i] = f2bf(in[i]);
}

// W[1024][1024] fp32 -> Wt[n][k] bf16 (transpose so B-fragments are contiguous)
__global__ __launch_bounds__(256) void transpose_f32_bf16_1024(
    const float* __restrict__ W, unsigned short* __restrict__ Wt) {
  int idx = blockIdx.x * blockDim.x + threadIdx.x;   // 1M elements
  int nrow = idx >> 10;
  int k    = idx & 1023;
  Wt[idx] = f2bf(W[k * 1024 + nrow]);
}

// ---------------------------------------------------------------------------
// C[M][N] = A[M][K] * B[K][N] with Bt = B^T stored [N][K] (bf16 in, fp32 acc).
// Block: 256 threads = 8 waves; block tile 32(M) x 512(N); wave tile 32x64
// (2 M-tiles x 4 N-tiles = 8 WMMAs per k-step -> 2x B reuse).
// The full 32x1024 A strip is staged in LDS ONCE (64KB of the WGP's 320KB),
// so the k-loop is barrier-free and the compiler's scheduler pipelines the
// clause'd B loads across tiles/iterations on its own (explicit register
// double-buffering was tried and regressed codegen: mov chains + hazard
// NOPs). K is fixed at 1024.
// ---------------------------------------------------------------------------
__global__ __launch_bounds__(256) void gemm_bf16_wmma(
    const unsigned short* __restrict__ A,
    const unsigned short* __restrict__ Bt,
    unsigned short* __restrict__ Cb,   // bf16 out (or null)
    float* __restrict__ Cf,            // fp32 out (or null)
    const float* __restrict__ bias,    // or null
    int M, int N) {
  __shared__ unsigned short As[32 * 1024];   // 64 KB

  const int wave  = threadIdx.x >> 5;
  const int lane  = threadIdx.x & 31;
  const int laneN = lane & 15;
  const int hlf   = lane >> 4;
  const int base  = hlf * 8;
  const int m0 = blockIdx.y * 32;
  const int n0 = blockIdx.x * 512 + wave * 64;

  // ---- stage the whole 32-row A strip (32 x 1024 bf16) into LDS ----
  for (int u = threadIdx.x; u < 4096; u += 256) {      // 4096 x 16B chunks
    const int r = u >> 7;            // 128 chunks per row
    const int c = (u & 127) * 8;     // half offset within row
    copy16_g2l(&A[(size_t)(m0 + r) * 1024 + c], &As[r * 1024 + c]);
  }
  wait_g2l();
  __syncthreads();

  v8f acc[2][4] = {};

  for (int k0 = 0; k0 < 1024; k0 += 32) {
    v16bf af0 = load_frag(&As[(laneN)      * 1024 + k0 + base]);
    v16bf af1 = load_frag(&As[(16 + laneN) * 1024 + k0 + base]);
#pragma unroll
    for (int t = 0; t < 4; ++t) {
      const unsigned short* bp =
          &Bt[(size_t)(n0 + t * 16 + laneN) * 1024 + k0 + base];
      __builtin_prefetch(bp + 64, 0, 3);   // next-next k-step, near cache
      v16bf bfr = load_frag(bp);
      acc[0][t] = WMMA_BF16(af0, bfr, acc[0][t]);
      acc[1][t] = WMMA_BF16(af1, bfr, acc[1][t]);
    }
  }

#pragma unroll
  for (int mt = 0; mt < 2; ++mt) {
#pragma unroll
    for (int t = 0; t < 4; ++t) {
#pragma unroll
      for (int v = 0; v < 8; ++v) {
        const int row = m0 + mt * 16 + v + hlf * 8;  // C layout: vgpr->M, lane->N
        const int col = n0 + t * 16 + laneN;
        const float val = acc[mt][t][v];
        if (Cf) Cf[(size_t)row * N + col] = val + (bias ? bias[col] : 0.0f);
        else    Cb[(size_t)row * N + col] = f2bf(val);
      }
    }
  }
  (void)M;
}

// ---------------------------------------------------------------------------
// Causal flash attention. Grid (S/64, H, B), 128 threads = 4 waves.
// Wave w owns query rows q0+16w..q0+16w+15 of head (b,h); keys processed in
// chunks of 32. K chunk and V^T chunk staged in LDS and shared by all waves.
// ---------------------------------------------------------------------------
__global__ __launch_bounds__(128) void flash_attn_wmma(
    const unsigned short* __restrict__ Qb,
    const unsigned short* __restrict__ Kb,
    const unsigned short* __restrict__ Vb,
    unsigned short* __restrict__ Ctx) {
  __shared__ unsigned short Kc[32 * 64];       // [key][dim]
  __shared__ unsigned short Vt[64 * 32];       // [dim][key]  (transposed)
  __shared__ unsigned short Pbuf[4][16 * 32];  // per-wave P relayout buffer

  const int b  = blockIdx.z;
  const int h  = blockIdx.y;
  const int q0 = blockIdx.x * 64;
  const int wave  = threadIdx.x >> 5;
  const int lane  = threadIdx.x & 31;
  const int laneN = lane & 15;
  const int hlf   = lane >> 4;
  const int base  = hlf * 8;
  const int tokBase = b * SS;
  const int hc = h * HD;

  // Preload the wave's Q fragments (head_dim 64 = two k-steps of 32).
  const unsigned short* qptr =
      Qb + (size_t)(tokBase + q0 + wave * 16 + laneN) * DD + hc;
  const v16bf qf0 = load_frag(qptr + base);
  const v16bf qf1 = load_frag(qptr + 32 + base);

  v8f o[4] = {};
  float m[8], l[8];
#pragma unroll
  for (int v = 0; v < 8; ++v) { m[v] = -3.0e38f; l[v] = 0.0f; }

  const int qw_max = q0 + wave * 16 + 15;

  for (int j0 = 0; j0 < q0 + 64; j0 += 32) {
    // ---- cooperative staging: K chunk row-major, V chunk transposed ----
    {
      const int tid = threadIdx.x;
      const int kr  = tid >> 2;          // key row 0..31
      const int seg = (tid & 3) * 16;    // dim segment
      const unsigned short* kp =
          Kb + (size_t)(tokBase + j0 + kr) * DD + hc + seg;
      copy16_g2l(kp,     &Kc[kr * 64 + seg]);
      copy16_g2l(kp + 8, &Kc[kr * 64 + seg + 8]);
      __builtin_prefetch(kp + 32 * DD, 0, 3);   // next key chunk

      const unsigned short* vp =
          Vb + (size_t)(tokBase + j0 + kr) * DD + hc + seg;
      __builtin_prefetch(vp + 32 * DD, 0, 3);
      union { uint4 q4[2]; unsigned short s[16]; } vd;
      vd.q4[0] = *(const uint4*)vp;
      vd.q4[1] = *(const uint4*)(vp + 8);
#pragma unroll
      for (int e = 0; e < 16; ++e) Vt[(seg + e) * 32 + kr] = vd.s[e];
    }
    wait_g2l();
    __syncthreads();

    if (j0 <= qw_max) {  // wave-uniform: EXEC stays all-ones for WMMA
      // ---- S = Q * K^T : two 16x16 score tiles ----
      v8f s0 = {}, s1 = {};
      s0 = WMMA_BF16(qf0, load_frag(&Kc[laneN * 64 + base]), s0);
      s1 = WMMA_BF16(qf0, load_frag(&Kc[(16 + laneN) * 64 + base]), s1);
      s0 = WMMA_BF16(qf1, load_frag(&Kc[laneN * 64 + 32 + base]), s0);
      s1 = WMMA_BF16(qf1, load_frag(&Kc[(16 + laneN) * 64 + 32 + base]), s1);

      // ---- causal mask + online softmax (row reductions across half-wave) --
      const int key0 = j0 + laneN;
      const int key1 = j0 + 16 + laneN;
      float p0[8], p1[8], alpha[8];
#pragma unroll
      for (int v = 0; v < 8; ++v) {
        const int qrow = q0 + wave * 16 + v + hlf * 8;
        float a0 = s0[v] * SM_SCALE; if (key0 > qrow) a0 = -3.0e38f;
        float a1 = s1[v] * SM_SCALE; if (key1 > qrow) a1 = -3.0e38f;
        float mx = fmaxf(a0, a1);
        mx = fmaxf(mx, __shfl_xor(mx, 1, 16));
        mx = fmaxf(mx, __shfl_xor(mx, 2, 16));
        mx = fmaxf(mx, __shfl_xor(mx, 4, 16));
        mx = fmaxf(mx, __shfl_xor(mx, 8, 16));
        const float mn = fmaxf(m[v], mx);
        alpha[v] = __expf(m[v] - mn);
        p0[v] = __expf(a0 - mn);
        p1[v] = __expf(a1 - mn);
        float rs = p0[v] + p1[v];
        rs += __shfl_xor(rs, 1, 16);
        rs += __shfl_xor(rs, 2, 16);
        rs += __shfl_xor(rs, 4, 16);
        rs += __shfl_xor(rs, 8, 16);
        l[v] = l[v] * alpha[v] + rs;
        m[v] = mn;
      }
#pragma unroll
      for (int t = 0; t < 4; ++t)
#pragma unroll
        for (int v = 0; v < 8; ++v) o[t][v] = o[t][v] * alpha[v];

      // ---- relayout P (C layout) -> bf16 A-fragment via LDS ----
#pragma unroll
      for (int v = 0; v < 8; ++v) {
        const int Mr = v + hlf * 8;
        Pbuf[wave][Mr * 32 + laneN]      = f2bf(p0[v]);
        Pbuf[wave][Mr * 32 + 16 + laneN] = f2bf(p1[v]);
      }
      asm volatile("s_wait_dscnt 0" ::: "memory");
      const v16bf pf = load_frag(&Pbuf[wave][laneN * 32 + base]);

      // ---- O += P * V ----
#pragma unroll
      for (int n = 0; n < 4; ++n) {
        v16bf vf = load_frag(&Vt[(n * 16 + laneN) * 32 + base]);
        o[n] = WMMA_BF16(pf, vf, o[n]);
      }
    }
    __syncthreads();
  }

  // ---- normalize and store ctx (bf16, [token][D] with head slice) ----
#pragma unroll
  for (int v = 0; v < 8; ++v) {
    const float inv = 1.0f / l[v];
    const int Mr = v + hlf * 8;
    const size_t rowoff = (size_t)(tokBase + q0 + wave * 16 + Mr) * DD + hc;
#pragma unroll
    for (int n = 0; n < 4; ++n)
      Ctx[rowoff + n * 16 + laneN] = f2bf(o[n][v] * inv);
  }
}

// ---------------------------------------------------------------------------
extern "C" void kernel_launch(void* const* d_in, const int* in_sizes, int n_in,
                              void* d_out, int out_size, void* d_ws,
                              size_t ws_size, hipStream_t stream) {
  (void)in_sizes; (void)n_in; (void)out_size; (void)ws_size;

  const float* x  = (const float*)d_in[0];
  const float* Wq = (const float*)d_in[1];
  const float* Wk = (const float*)d_in[2];
  const float* Wv = (const float*)d_in[3];
  const float* Wo = (const float*)d_in[4];
  const float* bo = (const float*)d_in[5];
  float* out = (float*)d_out;

  // workspace layout (bf16 elements); total 24*E = 48 MiB
  unsigned short* ws = (unsigned short*)d_ws;
  const size_t E = 1024 * 1024;
  unsigned short* xb  = ws;            // x bf16            : 4E
  unsigned short* WqT = ws + 4 * E;    // Wq^T bf16         : 1E
  unsigned short* WkT = ws + 5 * E;
  unsigned short* WvT = ws + 6 * E;
  unsigned short* WoT = ws + 7 * E;
  unsigned short* Qb  = ws + 8 * E;    // Q bf16            : 4E
  unsigned short* Kbp = ws + 12 * E;   // K bf16            : 4E
  unsigned short* Vbp = ws + 16 * E;   // V bf16            : 4E
  unsigned short* Cx  = ws + 20 * E;   // ctx bf16          : 4E

  cvt_f32_bf16<<<2048, 256, 0, stream>>>(x, xb, (int)(4 * E));
  transpose_f32_bf16_1024<<<4096, 256, 0, stream>>>(Wq, WqT);
  transpose_f32_bf16_1024<<<4096, 256, 0, stream>>>(Wk, WkT);
  transpose_f32_bf16_1024<<<4096, 256, 0, stream>>>(Wv, WvT);
  transpose_f32_bf16_1024<<<4096, 256, 0, stream>>>(Wo, WoT);

  const dim3 gg(2, 128);  // N/512 x M/32
  gemm_bf16_wmma<<<gg, 256, 0, stream>>>(xb, WqT, Qb, nullptr, nullptr,
                                         BB * SS, DD);
  gemm_bf16_wmma<<<gg, 256, 0, stream>>>(xb, WkT, Kbp, nullptr, nullptr,
                                         BB * SS, DD);
  gemm_bf16_wmma<<<gg, 256, 0, stream>>>(xb, WvT, Vbp, nullptr, nullptr,
                                         BB * SS, DD);

  const dim3 fg(SS / 64, HH, BB);
  flash_attn_wmma<<<fg, 128, 0, stream>>>(Qb, Kbp, Vbp, Cx);

  gemm_bf16_wmma<<<gg, 256, 0, stream>>>(Cx, WoT, nullptr, out, bo,
                                         BB * SS, DD);
}
